// SparseLayer_3410204033732
// MI455X (gfx1250) — matrix operand order repaired and checked by reading, not compiled, not run on hardware
//
#include <hip/hip_runtime.h>

// ---------------- problem constants (match reference) ----------------
#define BDIM   4096      // batch (M)
#define INDIM  9000      // K (unpadded)
#define KPAD   9024      // K padded to multiple of 32 (282 * 32)
#define NDIM   2048      // OUT_DIM (N)

// ---------------- GEMM tiling ----------------
#define BM 128
#define BN 128
#define BK 32
#define LDSROW 40        // padded LDS row stride in halfs (80 B, 16B-aligned)

typedef __attribute__((ext_vector_type(16))) _Float16 v16h;
typedef __attribute__((ext_vector_type(8)))  _Float16 v8h;
typedef __attribute__((ext_vector_type(8)))  float    v8f;
typedef __attribute__((ext_vector_type(4)))  float    v4f;
typedef __attribute__((ext_vector_type(4)))  unsigned int v4u;
typedef int v4i __attribute__((vector_size(16)));   // matches builtin param type

// ---------------- CDNA5 async global->LDS copy (guarded) ----------------
#if defined(__gfx1250__) && __has_builtin(__builtin_amdgcn_global_load_async_to_lds_b128)
#define USE_ASYNC 1
#else
#define USE_ASYNC 0
#endif

__device__ __forceinline__ void cp16(const _Float16* __restrict__ g, _Float16* l) {
#if USE_ASYNC
    // 16B DMA directly into LDS, tracked by ASYNCcnt (no VGPR staging)
    __builtin_amdgcn_global_load_async_to_lds_b128(
        (__attribute__((address_space(1))) v4i*)g,
        (__attribute__((address_space(3))) v4i*)l,
        0, 0);
#else
    v4u d = *(const v4u*)g;
    *(v4u*)l = d;
#endif
}

__device__ __forceinline__ void wait_async_copies() {
#if USE_ASYNC
#if __has_builtin(__builtin_amdgcn_s_wait_asynccnt)
    __builtin_amdgcn_s_wait_asynccnt(0);
#else
    asm volatile("s_wait_asynccnt 0x0" ::: "memory");
#endif
#endif
}

// ---------------- kernel 1: zero the f16 W^T workspace ----------------
__global__ void k_zero_w(v4u* __restrict__ w, unsigned int nvec) {
    unsigned int i = blockIdx.x * blockDim.x + threadIdx.x;
    if (i < nvec) {
        v4u z; z[0] = 0u; z[1] = 0u; z[2] = 0u; z[3] = 0u;
        w[i] = z;
    }
}

// ---------------- kernel 2: convert x fp32 -> f16, zero-pad K tail ----------------
__global__ void k_convert_x(const float* __restrict__ x, _Float16* __restrict__ x16) {
    // one thread handles 8 consecutive K elements; 9000 = 8*1125, KPAD/8 = 1128
    unsigned int id = blockIdx.x * blockDim.x + threadIdx.x;
    const unsigned int GPR = KPAD / 8;              // groups per row = 1128
    if (id >= (unsigned int)BDIM * GPR) return;
    unsigned int g   = id % GPR;
    unsigned int row = id / GPR;
    v8h h;
    if (g < INDIM / 8) {                            // fully valid group
        const v4f* p = (const v4f*)(x + (size_t)row * INDIM + (size_t)g * 8);
        v4f a = p[0], b = p[1];
#pragma unroll
        for (int i = 0; i < 4; ++i) { h[i] = (_Float16)a[i]; h[4 + i] = (_Float16)b[i]; }
    } else {                                        // padding groups 1125..1127
#pragma unroll
        for (int i = 0; i < 8; ++i) h[i] = (_Float16)0.f;
    }
    *(v8h*)(x16 + (size_t)row * KPAD + (size_t)g * 8) = h;
}

// ---------------- kernel 3: scatter-add sparse weights into f16 W^T ----------------
__device__ __forceinline__ void atomic_add_f16(_Float16* addr, float val) {
    unsigned int* base = (unsigned int*)((size_t)addr & ~(size_t)3);
    const bool hi = ((size_t)addr & 2) != 0;
    unsigned int old = *base;
    while (true) {
        unsigned short cur16 = hi ? (unsigned short)(old >> 16) : (unsigned short)(old & 0xffffu);
        _Float16 cur; __builtin_memcpy(&cur, &cur16, 2);
        _Float16 sum = (_Float16)((float)cur + val);
        unsigned short s16; __builtin_memcpy(&s16, &sum, 2);
        unsigned int neu = hi ? ((old & 0x0000ffffu) | ((unsigned int)s16 << 16))
                              : ((old & 0xffff0000u) | (unsigned int)s16);
        unsigned int prev = atomicCAS(base, old, neu);
        if (prev == old) break;
        old = prev;
    }
}

__global__ void k_scatter(const int* __restrict__ idx, const float* __restrict__ w,
                          _Float16* __restrict__ W16T, int nnz) {
    int i = blockIdx.x * blockDim.x + threadIdx.x;
    if (i >= nnz) return;
    int r = idx[2 * i];       // K index
    int c = idx[2 * i + 1];   // N index
    atomic_add_f16(W16T + (size_t)c * KPAD + r, w[i]);
}

// ---------------- kernel 4: double-buffered f16 WMMA GEMM + bias ----------------
__global__ __launch_bounds__(256) void k_gemm(const _Float16* __restrict__ X16,
                                              const _Float16* __restrict__ W16T,
                                              const float* __restrict__ bias,
                                              float* __restrict__ out) {
    __shared__ _Float16 sA[2][BM][LDSROW];   // A tile,  [m][k]
    __shared__ _Float16 sB[2][BN][LDSROW];   // B tile,  [n][k] (W already transposed)

    const int tid  = threadIdx.x;
    const int lane = tid & 31;
    const int wave = tid >> 5;            // 0..7
    const int wm   = wave >> 2;           // 0..1  (64 M-rows per wave-row)
    const int wn   = wave & 3;            // 0..3  (32 N-cols per wave-col)
    const int m0   = blockIdx.y * BM;
    const int n0   = blockIdx.x * BN;

    v8f acc[4][2];
#pragma unroll
    for (int s = 0; s < 4; ++s)
#pragma unroll
        for (int t = 0; t < 2; ++t)
#pragma unroll
            for (int r = 0; r < 8; ++r) acc[s][t][r] = 0.f;

    // tile loader mapping: each thread copies two 16B chunks of A and of B
    const int arow  = tid >> 1;           // 0..127
    const int apart = (tid & 1) * 16;     // 0 or 16 (halfs)
    const _Float16* gA = X16  + (size_t)(m0 + arow) * KPAD + apart;
    const _Float16* gB = W16T + (size_t)(n0 + arow) * KPAD + apart;

    const int mrow = lane & 15;
    const int half = lane >> 4;

    auto stage = [&](int buf, int k0) {
        cp16(gA + k0,     &sA[buf][arow][apart]);
        cp16(gA + k0 + 8, &sA[buf][arow][apart + 8]);
        cp16(gB + k0,     &sB[buf][arow][apart]);
        cp16(gB + k0 + 8, &sB[buf][arow][apart + 8]);
    };

    const int NT = KPAD / BK;             // 282 k-steps

    // prologue: tile 0 -> buffer 0
    stage(0, 0);
    wait_async_copies();
    __syncthreads();

    for (int kt = 0; kt < NT; ++kt) {
        const int cur = kt & 1;
        const bool has_next = (kt + 1) < NT;
        if (has_next) stage(cur ^ 1, (kt + 1) * BK);   // overlap copy with WMMA below

        // ---- LDS -> fragments (ISA 16-bit A 16x32 / B 32x16 layouts) ----
        v16h afr[4], bfr[2];
#pragma unroll
        for (int s = 0; s < 4; ++s) {
            const _Float16* pa = &sA[cur][wm * 64 + s * 16 + mrow][0];
            v8h lo = *(const v8h*)(pa + 8 * half);        // K = 8h .. 8h+7   -> v0..v3
            v8h hi = *(const v8h*)(pa + 16 + 8 * half);   // K = 16+8h ..+7   -> v4..v7
            afr[s] = __builtin_shufflevector(lo, hi, 0,1,2,3,4,5,6,7,8,9,10,11,12,13,14,15);
        }
#pragma unroll
        for (int t = 0; t < 2; ++t) {
            const _Float16* pb = &sB[cur][wn * 32 + t * 16 + mrow][0];
            v8h lo = *(const v8h*)(pb + 8 * half);        // K = 8h .. 8h+7
            v8h hi = *(const v8h*)(pb + 16 + 8 * half);   // K = 16+8h ..+7
            bfr[t] = __builtin_shufflevector(lo, hi, 0,1,2,3,4,5,6,7,8,9,10,11,12,13,14,15);
        }

        // ---- 8x v_wmma_f32_16x16x32_f16 per K-step ----
#pragma unroll
        for (int s = 0; s < 4; ++s)
#pragma unroll
            for (int t = 0; t < 2; ++t)
                acc[s][t] = __builtin_amdgcn_wmma_f32_16x16x32_f16(
                    false, afr[s], false, bfr[t], (short)0, acc[s][t], false, false);

        wait_async_copies();   // next tile resident in LDS before anyone passes barrier
        __syncthreads();
    }

    // ---- epilogue: C/D layout M = r + 8*half, N = lane&15; fuse bias ----
#pragma unroll
    for (int t = 0; t < 2; ++t) {
        const int n  = n0 + wn * 32 + t * 16 + mrow;
        const float bv = bias[n];
#pragma unroll
        for (int s = 0; s < 4; ++s) {
            const int mbase = m0 + wm * 64 + s * 16 + 8 * half;
#pragma unroll
            for (int r = 0; r < 8; ++r)
                out[(size_t)(mbase + r) * NDIM + n] = acc[s][t][r] + bv;
        }
    }
}

// ---------------- host launcher ----------------
extern "C" void kernel_launch(void* const* d_in, const int* in_sizes, int n_in,
                              void* d_out, int out_size, void* d_ws, size_t ws_size,
                              hipStream_t stream) {
    const float* x    = (const float*)d_in[0];
    const int*   idx  = (const int*)d_in[1];    // [NNZ,2] (row,col) pairs
    const float* wts  = (const float*)d_in[2];
    const float* bias = (const float*)d_in[3];
    float*       out  = (float*)d_out;
    const int    nnz  = in_sizes[2];

    char* ws = (char*)d_ws;
    _Float16* X16  = (_Float16*)ws;                                        // 4096*9024*2 = 73.9 MB
    _Float16* W16T = (_Float16*)(ws + (size_t)BDIM * KPAD * sizeof(_Float16)); // 2048*9024*2 = 37.0 MB

    // 1) zero f16 W^T (must happen every call: scatter accumulates)
    {
        unsigned int nvec = (unsigned int)((size_t)KPAD * NDIM / 8);       // 2,310,144 x 16B
        unsigned int blocks = (nvec + 255) / 256;
        k_zero_w<<<blocks, 256, 0, stream>>>((v4u*)W16T, nvec);
    }
    // 2) convert x to f16 with zero-padded K tail
    {
        unsigned int total = (unsigned int)BDIM * (KPAD / 8);              // 4,620,288
        unsigned int blocks = (total + 255) / 256;
        k_convert_x<<<blocks, 256, 0, stream>>>(x, X16);
    }
    // 3) scatter-add sparse weights directly into transposed f16 W
    k_scatter<<<(nnz + 255) / 256, 256, 0, stream>>>(idx, wts, W16T, nnz);

    // 4) WMMA GEMM + bias: grid = (N/BN, M/BM) = (16, 32), 256 threads
    dim3 grid(NDIM / BN, BDIM / BM);
    k_gemm<<<grid, 256, 0, stream>>>(X16, W16T, bias, out);
}
